// MultiHeadAttention_77799037599835
// MI455X (gfx1250) — matrix operand order repaired and checked by reading, not compiled, not run on hardware
//
#include <hip/hip_runtime.h>

// ---------------------------------------------------------------------------
// CDNA5 (gfx1250) transformer block: QKV proj -> flash attention -> out proj
// -> LN1 -> SiLU MLP -> LN2.
// Matmuls: v_wmma_f32_16x16x32_bf16.  GEMM tiles staged through LDS by the
// Tensor Data Mover (tensor_load_to_lds, TENSORcnt double-buffering).
// wave32 only.
// ---------------------------------------------------------------------------

typedef unsigned short u16;
typedef unsigned int   u32;
typedef unsigned long long u64;

typedef __attribute__((ext_vector_type(16))) __bf16 v16bf;
typedef __attribute__((ext_vector_type(8)))  float  v8f;
typedef __attribute__((ext_vector_type(4)))  unsigned int v4u;
typedef __attribute__((ext_vector_type(8)))  int v8i;
typedef __attribute__((ext_vector_type(4)))  int v4i;

#define SEQ   2048
#define EMB_  1024
#define HEADS 16
#define DHEAD 64
#define FFN   2048
#define NROWS 8192   // B*SEQ

#define FLAG_SILU 1
#define FLAG_VT   2

// LDS tile: 128 rows x 32 bf16, padded to 40 elems (80 B) per row so that
// per-lane b128 reads across 16 rows hit distinct bank groups.
#define LDS_STRIDE 40

#if __has_builtin(__builtin_amdgcn_tensor_load_to_lds)
#define HAVE_TDM 1
#else
#define HAVE_TDM 0
#endif

struct alignas(32) Frag {
  union { uint4 q[2]; v16bf v; };
};

__device__ __forceinline__ u16 f2bf(float f) {
  u32 u = __float_as_uint(f);
  u32 r = u + 0x7FFFu + ((u >> 16) & 1u);   // round-to-nearest-even
  return (u16)(r >> 16);
}

// A-matrix 16x32 bf16 fragment (contiguous-16 K mapping):
// lane: m = l&15, g = l>>4; e0..15 -> K = 16g+e
// caller passes p = row_base + 16*g
__device__ __forceinline__ void load_a(Frag& f, const u16* p) {
  f.q[0] = *reinterpret_cast<const uint4*>(p);
  f.q[1] = *reinterpret_cast<const uint4*>(p + 8);
}

// B-matrix 32x16 bf16 fragment (contiguous-16 K mapping):
// lane: n = l&15, g = l>>4; e0..15 -> K = 16g+e
// caller passes p = col_row_base + 16*g; 32 contiguous bytes
__device__ __forceinline__ void load_b(Frag& f, const u16* p) {
  f.q[0] = *reinterpret_cast<const uint4*>(p);
  f.q[1] = *reinterpret_cast<const uint4*>(p + 8);
}

__device__ __forceinline__ v8f wmma_bf16(const Frag& a, const Frag& b, v8f c) {
  return __builtin_amdgcn_wmma_f32_16x16x32_bf16(
      /*neg_a=*/false, a.v, /*neg_b=*/false, b.v,
      /*c_mod=*/(short)0, c, /*reuse_a=*/false, /*reuse_b=*/false);
}

#if HAVE_TDM
// Issue one TDM load of a 128x32 bf16 tile (row stride K elems) into LDS at
// byte offset lds_addr, padded 4 DWORDs every 16 DWORDs (-> 80 B row stride).
// D# packing per CDNA5 ISA 8.3/8.4.  2D tensor: groups 2/3 zero.
// This toolchain's builtin takes 6 args (g0 v4u, g1 v8i, g2 v4i, g3 v4i,
// extra v8i, cpol) -- extra group zero-filled.
__device__ __forceinline__ void tdm_load_tile(u32 lds_addr, const u16* gsrc,
                                              int kStride) {
  const u64 ga = (u64)(size_t)gsrc;
  v4u g0;
  g0[0] = 1u;                                   // count=1, user mode, no gather
  g0[1] = lds_addr;                             // lds_addr [63:32]
  g0[2] = (u32)ga;                              // global_addr [95:64]
  g0[3] = (u32)(ga >> 32) | (2u << 30);         // global_addr hi | type=2
  v8i g1;
  // data_size=1 (2B) | pad_enable | pad_interval=3 (16 DW) | pad_amount=3 (4 DW)
  g1[0] = (int)((1u << 16) | (1u << 20) | (3u << 22) | (3u << 25));
  g1[1] = (int)(32u << 16);                     // tensor_dim0[15:0]=32
  g1[2] = (int)(128u << 16);                    // tensor_dim0 hi=0 | tensor_dim1=128
  g1[3] = (int)(32u << 16);                     // tensor_dim1 hi=0 | tile_dim0=32
  g1[4] = 128;                                  // tile_dim1=128 | tile_dim2=0
  g1[5] = kStride;                              // tensor_dim0_stride lo (elems)
  g1[6] = 0;                                    // stride hi | dim1_stride lo
  g1[7] = 0;
  const v4i z4 = {0, 0, 0, 0};
  const v8i z8 = {0, 0, 0, 0, 0, 0, 0, 0};
  __builtin_amdgcn_tensor_load_to_lds(g0, g1, z4, z4, z8, 0);
}
#endif

// ---------------------------------------------------------------------------
// fp32 -> bf16 conversion
// ---------------------------------------------------------------------------
__global__ __launch_bounds__(256) void cvt_bf16(const float* __restrict__ s,
                                                u16* __restrict__ d, int n) {
  int i = blockIdx.x * 256 + threadIdx.x;
  int stride = gridDim.x * 256;
  for (; i < n; i += stride) d[i] = f2bf(s[i]);
}

// ---------------------------------------------------------------------------
// Generic GEMM:  C[m,n] = scale * sum_k A[m,k]*W[n,k]  (+bias, silu, resid)
// A: [M,K] bf16 row-major.  W: [N,K] bf16 row-major (torch Linear weight).
// Block = 128x128, 256 threads = 8 waves in 2(M)x4(N); wave = 4x2 wmma tiles.
// Tiles staged into LDS by the Tensor Data Mover, double buffered.
// ---------------------------------------------------------------------------
__global__ __launch_bounds__(256) void gemm_bf16(
    const u16* __restrict__ A, const u16* __restrict__ W,
    const float* __restrict__ bias, const float* __restrict__ resid,
    float* __restrict__ outF, u16* __restrict__ outB,
    int M, int N, int K, float scale, int flags) {
  const int lane = threadIdx.x & 31;
  const int wave = threadIdx.x >> 5;
  const int wm = wave >> 2;            // 0..1
  const int wn = wave & 3;             // 0..3
  const int g  = lane >> 4;            // half-wave
  const int ln = lane & 15;
  const int mBase = blockIdx.y * 128 + wm * 64;
  const int nBase = blockIdx.x * 128 + wn * 32;

  __shared__ u16 sA[2][128 * LDS_STRIDE];   // 10 KB each buffer
  __shared__ u16 sB[2][128 * LDS_STRIDE];

  const u16* aTile = A + (size_t)(blockIdx.y * 128) * K;   // + k
  const u16* bTile = W + (size_t)(blockIdx.x * 128) * K;   // + k

  const v8f zf = {0.f, 0.f, 0.f, 0.f, 0.f, 0.f, 0.f, 0.f};
  v8f acc[4][2];
#pragma unroll
  for (int i = 0; i < 4; ++i)
#pragma unroll
    for (int j = 0; j < 2; ++j) acc[i][j] = zf;

  const int kSteps = K >> 5;

#if HAVE_TDM
  if (wave == 0) {
    tdm_load_tile((u32)(size_t)&sA[0][0], aTile, K);
    tdm_load_tile((u32)(size_t)&sB[0][0], bTile, K);
  }
#else
  {
    // cooperative fallback: 256 thr, each moves 16 contiguous bf16 per tile
    const int row = threadIdx.x >> 1, c0 = (threadIdx.x & 1) * 16;
    uint4* da = reinterpret_cast<uint4*>(&sA[0][row * LDS_STRIDE + c0]);
    uint4* db = reinterpret_cast<uint4*>(&sB[0][row * LDS_STRIDE + c0]);
    const uint4* ga = reinterpret_cast<const uint4*>(aTile + (size_t)row * K + c0);
    const uint4* gb = reinterpret_cast<const uint4*>(bTile + (size_t)row * K + c0);
    da[0] = ga[0]; da[1] = ga[1];
    db[0] = gb[0]; db[1] = gb[1];
  }
#endif

  for (int s = 0; s < kSteps; ++s) {
    const int cur = s & 1;
#if HAVE_TDM
    if (wave == 0) {
      if (s + 1 < kSteps) {
        const int k = (s + 1) << 5;
        tdm_load_tile((u32)(size_t)&sA[cur ^ 1][0], aTile + k, K);
        tdm_load_tile((u32)(size_t)&sB[cur ^ 1][0], bTile + k, K);
        __builtin_amdgcn_s_wait_tensorcnt(2);   // current pair done, next in flight
      } else {
        __builtin_amdgcn_s_wait_tensorcnt(0);
      }
    }
#else
    if (s + 1 < kSteps) {
      const int k = (s + 1) << 5;
      const int row = threadIdx.x >> 1, c0 = (threadIdx.x & 1) * 16;
      uint4* da = reinterpret_cast<uint4*>(&sA[cur ^ 1][row * LDS_STRIDE + c0]);
      uint4* db = reinterpret_cast<uint4*>(&sB[cur ^ 1][row * LDS_STRIDE + c0]);
      const uint4* ga = reinterpret_cast<const uint4*>(aTile + (size_t)row * K + k + c0);
      const uint4* gb = reinterpret_cast<const uint4*>(bTile + (size_t)row * K + k + c0);
      da[0] = ga[0]; da[1] = ga[1];
      db[0] = gb[0]; db[1] = gb[1];
    }
#endif
    __syncthreads();                            // staged tile visible to all waves

    const u16* la = &sA[cur][0];
    const u16* lb = &sB[cur][0];
    Frag fa[4], fb[2];
#pragma unroll
    for (int i = 0; i < 4; ++i)
      load_a(fa[i], la + (wm * 64 + i * 16 + ln) * LDS_STRIDE + 16 * g);
#pragma unroll
    for (int j = 0; j < 2; ++j)
      load_b(fb[j], lb + (wn * 32 + j * 16 + ln) * LDS_STRIDE + 16 * g);
#pragma unroll
    for (int i = 0; i < 4; ++i)
#pragma unroll
      for (int j = 0; j < 2; ++j) acc[i][j] = wmma_bf16(fa[i], fb[j], acc[i][j]);
    __syncthreads();                            // reads done before buffer reuse
  }

#pragma unroll
  for (int j = 0; j < 2; ++j) {
    const int n = nBase + j * 16 + ln;
    const float bs = bias ? bias[n] : 0.f;
#pragma unroll
    for (int i = 0; i < 4; ++i) {
#pragma unroll
      for (int r = 0; r < 8; ++r) {
        const int m = mBase + i * 16 + r + 8 * g;
        float v = acc[i][j][r] * scale + bs;
        if (flags & FLAG_SILU) v = v * (1.f / (1.f + __expf(-v)));
        if (resid) v += resid[(size_t)m * N + n];
        if (outF) outF[(size_t)m * N + n] = v;
        if (outB) {
          size_t idx;
          if (flags & FLAG_VT) {                // store V transposed: [b, n, key]
            const int b = m >> 11;              // SEQ = 2048
            const int key = m & (SEQ - 1);
            idx = ((size_t)(b * N + n)) * SEQ + key;
          } else {
            idx = (size_t)m * N + n;
          }
          outB[idx] = f2bf(v);
        }
      }
    }
  }
}

// ---------------------------------------------------------------------------
// Flash attention: grid = (SEQ/128, B*HEADS), 256 thr = 8 waves.
// Wave owns 16 q-rows; loops keys in chunks of 32 with online softmax.
// Q already carries 1/sqrt(D).  Vt layout: [(b*EMB + h*64 + d), key].
// ---------------------------------------------------------------------------
__global__ __launch_bounds__(256) void flash_attn(
    const u16* __restrict__ Q, const u16* __restrict__ Kc,
    const u16* __restrict__ Vt, u16* __restrict__ O) {
  const int lane = threadIdx.x & 31;
  const int wave = threadIdx.x >> 5;
  const int g  = lane >> 4;
  const int ln = lane & 15;
  const int bh = blockIdx.y;
  const int b  = bh >> 4;
  const int h  = bh & 15;
  const int q0 = blockIdx.x * 128 + wave * 16;

  __shared__ u16 sP[8][16 * 32];     // per-wave P tile (16 q x 32 keys)
  u16* myP = sP[wave];

  const u16* qrow = Q + (size_t)(b * SEQ + q0 + ln) * EMB_ + h * DHEAD;
  Frag fq[2];
  load_a(fq[0], qrow + 0  + 16 * g);
  load_a(fq[1], qrow + 32 + 16 * g);

  const v8f zf = {0.f, 0.f, 0.f, 0.f, 0.f, 0.f, 0.f, 0.f};
  v8f acc[4];
#pragma unroll
  for (int t = 0; t < 4; ++t) acc[t] = zf;
  float mr[8], lr[8];
#pragma unroll
  for (int r = 0; r < 8; ++r) { mr[r] = -1e30f; lr[r] = 0.f; }

  const u16* kbase = Kc + (size_t)(b * SEQ) * EMB_ + h * DHEAD;
  const u16* vbase = Vt + ((size_t)(b * EMB_ + h * DHEAD)) * SEQ;

  for (int j = 0; j < SEQ; j += 32) {
    v8f s0 = zf, s1 = zf;
#pragma unroll
    for (int c = 0; c < 2; ++c) {           // contraction over d in 2 chunks
      Frag fk0, fk1;
      const u16* kp = kbase + (size_t)(j + ln) * EMB_ + 32 * c + 16 * g;
      load_b(fk0, kp);
      load_b(fk1, kp + (size_t)16 * EMB_);
      s0 = wmma_bf16(fq[c], fk0, s0);
      s1 = wmma_bf16(fq[c], fk1, s1);
    }
    // online softmax over these 32 key-columns
#pragma unroll
    for (int r = 0; r < 8; ++r) {
      float smax = fmaxf(s0[r], s1[r]);
#pragma unroll
      for (int off = 1; off < 16; off <<= 1)
        smax = fmaxf(smax, __shfl_xor(smax, off, 32));
      const float mn = fmaxf(mr[r], smax);
      const float al = __expf(mr[r] - mn);
      mr[r] = mn;
      const float p0 = __expf(s0[r] - mn);
      const float p1 = __expf(s1[r] - mn);
      float rs = p0 + p1;
#pragma unroll
      for (int off = 1; off < 16; off <<= 1) rs += __shfl_xor(rs, off, 32);
      lr[r] = lr[r] * al + rs;
      acc[0][r] *= al; acc[1][r] *= al; acc[2][r] *= al; acc[3][r] *= al;
      const int mrow = r + 8 * g;           // C-layout row
      myP[mrow * 32 + ln]      = f2bf(p0);
      myP[mrow * 32 + 16 + ln] = f2bf(p1);
    }
    __syncthreads();                        // P visible (C-frag -> A-frag relayout)
    Frag fp;
    load_a(fp, myP + ln * 32 + 16 * g);
    __syncthreads();                        // reads done before next-iter writes
#pragma unroll
    for (int t = 0; t < 4; ++t) {
      Frag fv;
      load_b(fv, vbase + (size_t)(16 * t + ln) * SEQ + j + 16 * g);
      acc[t] = wmma_bf16(fp, fv, acc[t]);
    }
  }

#pragma unroll
  for (int t = 0; t < 4; ++t)
#pragma unroll
    for (int r = 0; r < 8; ++r) {
      const int m   = q0 + r + 8 * g;
      const int col = h * DHEAD + 16 * t + ln;
      O[(size_t)(b * SEQ + m) * EMB_ + col] = f2bf(acc[t][r] / lr[r]);
    }
}

// ---------------------------------------------------------------------------
// LayerNorm over last dim (C = 1024), one block per row.
// ---------------------------------------------------------------------------
__global__ __launch_bounds__(256) void layernorm_k(
    const float* __restrict__ X, const float* __restrict__ gw,
    const float* __restrict__ bw, float* __restrict__ outF,
    u16* __restrict__ outB, int C) {
  const int row = blockIdx.x;
  const int tid = threadIdx.x;
  const float* x = X + (size_t)row * C;
  float v[4], s = 0.f, s2 = 0.f;
#pragma unroll
  for (int i = 0; i < 4; ++i) {
    v[i] = x[tid + i * 256];
    s += v[i];
    s2 += v[i] * v[i];
  }
  __shared__ float rs[256], rs2[256];
  rs[tid] = s; rs2[tid] = s2;
  __syncthreads();
  for (int off = 128; off > 0; off >>= 1) {
    if (tid < off) { rs[tid] += rs[tid + off]; rs2[tid] += rs2[tid + off]; }
    __syncthreads();
  }
  const float mean = rs[0] / (float)C;
  const float var  = rs2[0] / (float)C - mean * mean;
  const float inv  = rsqrtf(var + 1e-5f);
#pragma unroll
  for (int i = 0; i < 4; ++i) {
    const int c = tid + i * 256;
    const float y = (v[i] - mean) * inv * gw[c] + bw[c];
    if (outF) outF[(size_t)row * C + c] = y;
    if (outB) outB[(size_t)row * C + c] = f2bf(y);
  }
}

// ---------------------------------------------------------------------------
// Host launcher
// ---------------------------------------------------------------------------
extern "C" void kernel_launch(void* const* d_in, const int* in_sizes, int n_in,
                              void* d_out, int out_size, void* d_ws, size_t ws_size,
                              hipStream_t stream) {
  (void)in_sizes; (void)n_in; (void)out_size; (void)ws_size;
  const float* x      = (const float*)d_in[0];
  const float* keys   = (const float*)d_in[1];
  const float* Wq     = (const float*)d_in[2];
  const float* Wk     = (const float*)d_in[3];
  const float* Wv     = (const float*)d_in[4];
  const float* Wo     = (const float*)d_in[5];
  const float* ln1_g  = (const float*)d_in[6];
  const float* ln1_b  = (const float*)d_in[7];
  const float* fc1_w  = (const float*)d_in[8];
  const float* fc1_b  = (const float*)d_in[9];
  const float* fc2_w  = (const float*)d_in[10];
  const float* fc2_b  = (const float*)d_in[11];
  const float* ln2_g  = (const float*)d_in[12];
  const float* ln2_b  = (const float*)d_in[13];
  float* out = (float*)d_out;

  char* ws = (char*)d_ws;
  const size_t MB = 1ull << 20;
  u16* xb   = (u16*)(ws + 0 * MB);    // 16 MB
  u16* kb   = (u16*)(ws + 16 * MB);   // 16 MB
  u16* wqb  = (u16*)(ws + 32 * MB);   // 2 MB each
  u16* wkb  = (u16*)(ws + 34 * MB);
  u16* wvb  = (u16*)(ws + 36 * MB);
  u16* wob  = (u16*)(ws + 38 * MB);
  u16* f1b  = (u16*)(ws + 40 * MB);   // 4 MB
  u16* f2b  = (u16*)(ws + 44 * MB);   // 4 MB
  u16* Qb   = (u16*)(ws + 48 * MB);   // 16 MB
  u16* Kb   = (u16*)(ws + 64 * MB);   // 16 MB
  u16* Vt   = (u16*)(ws + 80 * MB);   // 16 MB
  u16* attb = (u16*)(ws + 96 * MB);   // 16 MB
  float* y1     = (float*)(ws + 112 * MB);  // 32 MB (reused for ffn out)
  float* interf = (float*)(ws + 144 * MB);  // 32 MB
  u16* interb   = (u16*)(ws + 176 * MB);    // 16 MB
  u16* hb       = (u16*)(ws + 48 * MB);     // 32 MB alias over Qb+Kb (dead)

  const int ACT = NROWS * EMB_;   // 8388608
  const int WSZ = EMB_ * EMB_;    // 1048576
  const int FSZ = FFN * EMB_;     // 2097152

  // 1) convert activations + weights to bf16
  cvt_bf16<<<8192, 256, 0, stream>>>(x,     xb,  ACT);
  cvt_bf16<<<8192, 256, 0, stream>>>(keys,  kb,  ACT);
  cvt_bf16<<<2048, 256, 0, stream>>>(Wq,    wqb, WSZ);
  cvt_bf16<<<2048, 256, 0, stream>>>(Wk,    wkb, WSZ);
  cvt_bf16<<<2048, 256, 0, stream>>>(Wv,    wvb, WSZ);
  cvt_bf16<<<2048, 256, 0, stream>>>(Wo,    wob, WSZ);
  cvt_bf16<<<4096, 256, 0, stream>>>(fc1_w, f1b, FSZ);
  cvt_bf16<<<4096, 256, 0, stream>>>(fc2_w, f2b, FSZ);

  // 2) projections.  1/sqrt(64) folded into Q.  V stored transposed.
  dim3 gProj(EMB_ / 128, NROWS / 128);
  gemm_bf16<<<gProj, 256, 0, stream>>>(xb, wqb, nullptr, nullptr, nullptr, Qb,
                                       NROWS, EMB_, EMB_, 0.125f, 0);
  gemm_bf16<<<gProj, 256, 0, stream>>>(kb, wkb, nullptr, nullptr, nullptr, Kb,
                                       NROWS, EMB_, EMB_, 1.0f, 0);
  gemm_bf16<<<gProj, 256, 0, stream>>>(kb, wvb, nullptr, nullptr, nullptr, Vt,
                                       NROWS, EMB_, EMB_, 1.0f, FLAG_VT);

  // 3) attention
  dim3 gAtt(SEQ / 128, 4 * HEADS);
  flash_attn<<<gAtt, 256, 0, stream>>>(Qb, Kb, Vt, attb);

  // 4) output projection + residual(x)
  gemm_bf16<<<gProj, 256, 0, stream>>>(attb, wob, nullptr, x, y1, nullptr,
                                       NROWS, EMB_, EMB_, 1.0f, 0);

  // 5) LN1 -> inter (f32 + bf16)
  layernorm_k<<<NROWS, 256, 0, stream>>>(y1, ln1_g, ln1_b, interf, interb, EMB_);

  // 6) fc1 + SiLU
  dim3 gFc1(FFN / 128, NROWS / 128);
  gemm_bf16<<<gFc1, 256, 0, stream>>>(interb, f1b, fc1_b, nullptr, nullptr, hb,
                                      NROWS, FFN, EMB_, 1.0f, FLAG_SILU);

  // 7) fc2 + residual(inter)
  gemm_bf16<<<gProj, 256, 0, stream>>>(hb, f2b, fc2_b, interf, y1, nullptr,
                                       NROWS, EMB_, FFN, 1.0f, 0);

  // 8) LN2 -> out
  layernorm_k<<<NROWS, 256, 0, stream>>>(y1, ln2_g, ln2_b, out, nullptr, EMB_);
}
